// DifferentiableSelectCopy_37649683316888
// MI455X (gfx1250) — matrix operand order repaired and checked by reading, not compiled, not run on hardware
//
#include <hip/hip_runtime.h>

typedef __attribute__((ext_vector_type(16))) _Float16 v16h;
typedef __attribute__((ext_vector_type(8)))  float    v8f;

constexpr int B = 64, M = 16384, D = 64, C = 256, S = 3;
constexpr float EPS = 1e-8f;
constexpr int K2_BLKS = M / 128;   // 128 row-chunks per batch

// ---- workspace layout (float offsets) ----
constexpr size_t OFF_QN_R  = 0;
constexpr size_t OFF_QN_W  = OFF_QN_R  + (size_t)B * D;
constexpr size_t OFF_ERASE = OFF_QN_W  + (size_t)B * D;
constexpr size_t OFF_ADD   = OFF_ERASE + (size_t)B * D;
constexpr size_t OFF_SCAL  = OFF_ADD   + (size_t)B * D;          // B*16 per-batch scalars
constexpr size_t OFF_SS    = OFF_SCAL  + (size_t)B * 16;         // B*2 sharpen sums
constexpr size_t OFF_PMAX  = OFF_SS    + (size_t)B * 2;          // 2*B*K2_BLKS partial maxes
constexpr size_t OFF_SIM_R = OFF_PMAX  + (size_t)2 * B * K2_BLKS;
constexpr size_t OFF_SIM_W = OFF_SIM_R + (size_t)B * M;
constexpr size_t OFF_RPART = OFF_SIM_W + (size_t)B * M;          // B*K2_BLKS*D read partials
// scal slots: 0 beta_r,1 beta_w,2 gate_r,3 gate_w,4 gam_r,5 gam_w,
//             6-8 shift_r, 9-11 shift_w, 12 max_r,13 max_w,14 invden_r,15 invden_w

__device__ __forceinline__ float softplusf(float x) {
  return fmaxf(x, 0.f) + log1pf(expf(-fabsf(x)));
}

// unnormalized sharpened weight for one row m (content -> gate -> shift -> pow)
__device__ __forceinline__ float head_sharp(const float* __restrict__ sim,
                                            const float* __restrict__ initv,
                                            int m, float beta, float mx, float invden,
                                            float gate, float s0, float s1, float s2,
                                            float gamma) {
  int mm = (m == 0) ? (M - 1) : (m - 1);
  int mp = (m == M - 1) ? 0 : (m + 1);
  float cm = expf(beta * (sim[mm] - mx)) * invden;
  float c0 = expf(beta * (sim[m]  - mx)) * invden;
  float cp = expf(beta * (sim[mp] - mx)) * invden;
  float og = 1.f - gate;
  float shifted = s0 * (gate * cm + og * initv[mm])
                + s1 * (gate * c0 + og * initv[m])
                + s2 * (gate * cp + og * initv[mp]);
  return powf(shifted + EPS, gamma);
}

__device__ __forceinline__ float blockReduceSumF(float v, float* sbuf) {
  int tid = threadIdx.x;
  sbuf[tid] = v; __syncthreads();
  for (int s = 128; s > 0; s >>= 1) {
    if (tid < s) sbuf[tid] += sbuf[tid + s];
    __syncthreads();
  }
  float r = sbuf[0]; __syncthreads();
  return r;
}

// ---------------- K1: controller projections ----------------
__global__ __launch_bounds__(256) void k1_small(
    const float* __restrict__ ctrl,
    const float* __restrict__ Wk_r, const float* __restrict__ Wb_r, const float* __restrict__ bb_r,
    const float* __restrict__ Wg_r, const float* __restrict__ bg_r,
    const float* __restrict__ Ws_r, const float* __restrict__ bs_r,
    const float* __restrict__ Wgam_r, const float* __restrict__ bgam_r,
    const float* __restrict__ Wk_w, const float* __restrict__ Wb_w, const float* __restrict__ bb_w,
    const float* __restrict__ Wg_w, const float* __restrict__ bg_w,
    const float* __restrict__ Ws_w, const float* __restrict__ bs_w,
    const float* __restrict__ Wgam_w, const float* __restrict__ bgam_w,
    const float* __restrict__ We, const float* __restrict__ be,
    const float* __restrict__ Wa, const float* __restrict__ ba,
    float* __restrict__ ws) {
  const int b = blockIdx.x, tid = threadIdx.x;
  __shared__ float sc[C];
  __shared__ float skey[2][D];
  __shared__ float sdots[12];
  __shared__ float snorm[2];
  sc[tid] = ctrl[(size_t)b * C + tid];
  __syncthreads();

  if (tid < 64) {
    int d = tid; float a = 0.f;
    for (int k = 0; k < C; ++k) a += sc[k] * Wk_r[k * D + d];
    skey[0][d] = a;
  } else if (tid < 128) {
    int d = tid - 64; float a = 0.f;
    for (int k = 0; k < C; ++k) a += sc[k] * Wk_w[k * D + d];
    skey[1][d] = a;
  } else if (tid < 192) {
    int d = tid - 128; float a = 0.f;
    for (int k = 0; k < C; ++k) a += sc[k] * We[k * D + d];
    a += be[d];
    ws[OFF_ERASE + (size_t)b * D + d] = 1.f / (1.f + expf(-a));
  } else {
    int d = tid - 192; float a = 0.f;
    for (int k = 0; k < C; ++k) a += sc[k] * Wa[k * D + d];
    a += ba[d];
    ws[OFF_ADD + (size_t)b * D + d] = tanhf(a);
  }
  // 12 scalar projections: 0 Wb_r, 1 Wg_r, 2 Wgam_r, 3-5 Ws_r cols, 6 Wb_w, 7 Wg_w, 8 Wgam_w, 9-11 Ws_w
  if (tid < 12) {
    const float* Wp; int stride = 1, colo = 0;
    switch (tid) {
      case 0: Wp = Wb_r; break;   case 1: Wp = Wg_r; break;   case 2: Wp = Wgam_r; break;
      case 3: case 4: case 5: Wp = Ws_r; stride = S; colo = tid - 3; break;
      case 6: Wp = Wb_w; break;   case 7: Wp = Wg_w; break;   case 8: Wp = Wgam_w; break;
      default: Wp = Ws_w; stride = S; colo = tid - 9; break;
    }
    float a = 0.f;
    for (int k = 0; k < C; ++k) a += sc[k] * Wp[k * stride + colo];
    sdots[tid] = a;
  }
  __syncthreads();

  if (tid < 2) {
    float a = 0.f;
    for (int d = 0; d < D; ++d) { float v = skey[tid][d]; a += v * v; }
    snorm[tid] = sqrtf(a);
  }
  if (tid == 2) {
    float* sl = ws + OFF_SCAL + (size_t)b * 16;
    sl[0] = softplusf(sdots[0] + bb_r[0]) + 1.f;
    sl[1] = softplusf(sdots[6] + bb_w[0]) + 1.f;
    sl[2] = 1.f / (1.f + expf(-(sdots[1] + bg_r[0])));
    sl[3] = 1.f / (1.f + expf(-(sdots[7] + bg_w[0])));
    sl[4] = softplusf(sdots[2] + bgam_r[0]) + 1.f;
    sl[5] = softplusf(sdots[8] + bgam_w[0]) + 1.f;
    {
      float p0 = sdots[3] + bs_r[0], p1 = sdots[4] + bs_r[1], p2 = sdots[5] + bs_r[2];
      float mx = fmaxf(p0, fmaxf(p1, p2));
      float e0 = expf(p0 - mx), e1 = expf(p1 - mx), e2 = expf(p2 - mx);
      float inv = 1.f / (e0 + e1 + e2);
      sl[6] = e0 * inv; sl[7] = e1 * inv; sl[8] = e2 * inv;
    }
    {
      float p0 = sdots[9] + bs_w[0], p1 = sdots[10] + bs_w[1], p2 = sdots[11] + bs_w[2];
      float mx = fmaxf(p0, fmaxf(p1, p2));
      float e0 = expf(p0 - mx), e1 = expf(p1 - mx), e2 = expf(p2 - mx);
      float inv = 1.f / (e0 + e1 + e2);
      sl[9] = e0 * inv; sl[10] = e1 * inv; sl[11] = e2 * inv;
    }
  }
  __syncthreads();
  if (tid < 128) {
    int h = tid >> 6, d = tid & 63;
    ws[(h ? OFF_QN_W : OFF_QN_R) + (size_t)b * D + d] = skey[h][d] / (snorm[h] + EPS);
  }
}

// ---------------- K2: cosine similarities via WMMA ----------------
__global__ __launch_bounds__(256) void k2_sim(const float* __restrict__ mem,
                                              float* __restrict__ ws) {
  const int b = blockIdx.y, tid = threadIdx.x;
  const int wave = tid >> 5, lane = tid & 31;
  const int laneLo = lane & 15, laneHi = lane >> 4;
  const int r0 = blockIdx.x * 128 + wave * 16;
  const int row = r0 + laneLo;
  const float* rp = mem + ((size_t)b * M + row) * D;
  const int base = laneHi * 8;

  // Per CDNA5 16-bit A 16x32 layout: this lane holds row (r0+laneLo),
  // K subsets {base..base+7, base+16..base+23} per 32-wide K chunk.
  float seg[4][8];
#pragma unroll
  for (int s = 0; s < 4; ++s) {
    const float4* p = reinterpret_cast<const float4*>(rp + base + s * 16);
    float4 x = p[0], y = p[1];
    seg[s][0] = x.x; seg[s][1] = x.y; seg[s][2] = x.z; seg[s][3] = x.w;
    seg[s][4] = y.x; seg[s][5] = y.y; seg[s][6] = y.z; seg[s][7] = y.w;
  }
  float ssq = 0.f;
#pragma unroll
  for (int s = 0; s < 4; ++s)
#pragma unroll
    for (int e = 0; e < 8; ++e) ssq += seg[s][e] * seg[s][e];
  float normSq = ssq + __shfl_xor(ssq, 16, 32);  // full f32 row norm^2

  v16h a0, a1;
#pragma unroll
  for (int e = 0; e < 8; ++e) {
    a0[e]     = (_Float16)seg[0][e];
    a0[8 + e] = (_Float16)seg[1][e];
    a1[e]     = (_Float16)seg[2][e];
    a1[8 + e] = (_Float16)seg[3][e];
  }
  // B-matrix 32x16: lane = N column; lane half h holds K {8h..8h+7, 16+8h..16+8h+7}. Col0 = qn_r, col1 = qn_w.
  const float* qnr = ws + OFF_QN_R + (size_t)b * D;
  const float* qnw = ws + OFF_QN_W + (size_t)b * D;
  const int col = laneLo, khl = laneHi * 8;
  const float* q = (col == 0) ? qnr : ((col == 1) ? qnw : nullptr);
  v16h b0, b1;
#pragma unroll
  for (int e = 0; e < 16; ++e) {
    b0[e] = q ? (_Float16)q[khl + (e & 7) + ((e >> 3) << 4)]      : (_Float16)0.f;
    b1[e] = q ? (_Float16)q[32 + khl + (e & 7) + ((e >> 3) << 4)] : (_Float16)0.f;
  }
  v8f c = {0.f, 0.f, 0.f, 0.f, 0.f, 0.f, 0.f, 0.f};
  c = __builtin_amdgcn_wmma_f32_16x16x32_f16(false, a0, false, b0, (short)0, c, false, false);
  c = __builtin_amdgcn_wmma_f32_16x16x32_f16(false, a1, false, b1, (short)0, c, false, false);

  // D layout: lane<16 -> N=lane, c[i]=row i; lane>=16 -> N=lane-16, c[i]=row 8+i
  __shared__ float sdot[8][2][16];
  if (col < 2) {
    int rbase = laneHi * 8;
#pragma unroll
    for (int i = 0; i < 8; ++i) sdot[wave][col][rbase + i] = c[i];
  }
  __syncthreads();

  const int h = laneHi, j = laneLo;
  float sim = sdot[wave][h][j] / (sqrtf(normSq) + EPS);
  float* simArr = ws + (h ? OFF_SIM_W : OFF_SIM_R);
  simArr[(size_t)b * M + r0 + j] = sim;

  __shared__ float smax[256];
  smax[tid] = sim;
  __syncthreads();
  if (tid == 0) {
    float m0 = -3.4e38f, m1 = -3.4e38f;
    for (int i = 0; i < 256; ++i) {
      if (((i >> 4) & 1) == 0) m0 = fmaxf(m0, smax[i]);
      else                     m1 = fmaxf(m1, smax[i]);
    }
    ws[OFF_PMAX + (size_t)b * K2_BLKS + blockIdx.x] = m0;
    ws[OFF_PMAX + (size_t)B * K2_BLKS + (size_t)b * K2_BLKS + blockIdx.x] = m1;
  }
}

// ---------------- K3: softmax stats + sharpen sums ----------------
__global__ __launch_bounds__(256) void k3_stats(const float* __restrict__ init_r,
                                                const float* __restrict__ init_w,
                                                float* __restrict__ ws) {
  const int b = blockIdx.x, tid = threadIdx.x;
  __shared__ float sbuf[256];
  const float* simR = ws + OFF_SIM_R + (size_t)b * M;
  const float* simW = ws + OFF_SIM_W + (size_t)b * M;
  float* sl = ws + OFF_SCAL + (size_t)b * 16;

  float v0 = (tid < K2_BLKS) ? ws[OFF_PMAX + (size_t)b * K2_BLKS + tid] : -3.4e38f;
  float v1 = (tid < K2_BLKS) ? ws[OFF_PMAX + (size_t)B * K2_BLKS + (size_t)b * K2_BLKS + tid] : -3.4e38f;
  sbuf[tid] = v0; __syncthreads();
  for (int s = 128; s > 0; s >>= 1) { if (tid < s) sbuf[tid] = fmaxf(sbuf[tid], sbuf[tid + s]); __syncthreads(); }
  float mx0 = sbuf[0]; __syncthreads();
  sbuf[tid] = v1; __syncthreads();
  for (int s = 128; s > 0; s >>= 1) { if (tid < s) sbuf[tid] = fmaxf(sbuf[tid], sbuf[tid + s]); __syncthreads(); }
  float mx1 = sbuf[0]; __syncthreads();

  float beta0 = sl[0], beta1 = sl[1];
  float a0 = 0.f, a1 = 0.f;
  for (int m = tid; m < M; m += 256) {
    a0 += expf(beta0 * (simR[m] - mx0));
    a1 += expf(beta1 * (simW[m] - mx1));
  }
  float den0 = blockReduceSumF(a0, sbuf);
  float den1 = blockReduceSumF(a1, sbuf);
  float inv0 = 1.f / den0, inv1 = 1.f / den1;
  if (tid == 0) { sl[12] = mx0; sl[13] = mx1; sl[14] = inv0; sl[15] = inv1; }

  float g0 = sl[2], g1 = sl[3], gm0 = sl[4], gm1 = sl[5];
  float s00 = sl[6], s01 = sl[7], s02 = sl[8];
  float s10 = sl[9], s11 = sl[10], s12 = sl[11];
  float acc0 = 0.f, acc1 = 0.f;
  for (int m = tid; m < M; m += 256) {
    acc0 += head_sharp(simR, init_r, m, beta0, mx0, inv0, g0, s00, s01, s02, gm0);
    acc1 += head_sharp(simW, init_w, m, beta1, mx1, inv1, g1, s10, s11, s12, gm1);
  }
  float t0 = blockReduceSumF(acc0, sbuf);
  float t1 = blockReduceSumF(acc1, sbuf);
  if (tid == 0) { ws[OFF_SS + (size_t)b * 2 + 0] = t0; ws[OFF_SS + (size_t)b * 2 + 1] = t1; }
}

// ---------------- K5: erase/add update + read partials ----------------
__global__ __launch_bounds__(256) void k5_final(const float* __restrict__ mem,
                                                const float* __restrict__ init_r,
                                                const float* __restrict__ init_w,
                                                float* __restrict__ ws,
                                                float* __restrict__ out_mem) {
  const int b = blockIdx.y, chunk = blockIdx.x, tid = threadIdx.x;
  const int r0 = chunk * 128;
  const float* sl   = ws + OFF_SCAL + (size_t)b * 16;
  const float* simR = ws + OFF_SIM_R + (size_t)b * M;
  const float* simW = ws + OFF_SIM_W + (size_t)b * M;

  __shared__ float swr[128], sww[128];
  if (tid < 128) {
    int m = r0 + tid;
    float ur = head_sharp(simR, init_r, m, sl[0], sl[12], sl[14], sl[2], sl[6], sl[7], sl[8], sl[4]);
    float uw = head_sharp(simW, init_w, m, sl[1], sl[13], sl[15], sl[3], sl[9], sl[10], sl[11], sl[5]);
    swr[tid] = ur / (ws[OFF_SS + (size_t)b * 2 + 0] + EPS);
    sww[tid] = uw / (ws[OFF_SS + (size_t)b * 2 + 1] + EPS);
  }
  __syncthreads();

  const int cg = tid & 3, col0 = cg * 16;
  float er[16], ad[16];
  {
    const float4* ep = reinterpret_cast<const float4*>(ws + OFF_ERASE + (size_t)b * D + col0);
    const float4* ap = reinterpret_cast<const float4*>(ws + OFF_ADD   + (size_t)b * D + col0);
#pragma unroll
    for (int i = 0; i < 4; ++i) {
      float4 e = ep[i], a = ap[i];
      er[i * 4 + 0] = e.x; er[i * 4 + 1] = e.y; er[i * 4 + 2] = e.z; er[i * 4 + 3] = e.w;
      ad[i * 4 + 0] = a.x; ad[i * 4 + 1] = a.y; ad[i * 4 + 2] = a.z; ad[i * 4 + 3] = a.w;
    }
  }
  float acc[16];
#pragma unroll
  for (int i = 0; i < 16; ++i) acc[i] = 0.f;

  for (int p = 0; p < 2; ++p) {
    int rl = p * 64 + (tid >> 2);
    int m = r0 + rl;
    float wrm = swr[rl], wwm = sww[rl];
    const float4* mp = reinterpret_cast<const float4*>(mem     + ((size_t)b * M + m) * D + col0);
    float4*       op = reinterpret_cast<float4*>      (out_mem + ((size_t)b * M + m) * D + col0);
#pragma unroll
    for (int i = 0; i < 4; ++i) {
      float4 v = mp[i];
      float4 o;
      o.x = v.x * (1.f - wwm * er[i * 4 + 0]) + wwm * ad[i * 4 + 0];
      o.y = v.y * (1.f - wwm * er[i * 4 + 1]) + wwm * ad[i * 4 + 1];
      o.z = v.z * (1.f - wwm * er[i * 4 + 2]) + wwm * ad[i * 4 + 2];
      o.w = v.w * (1.f - wwm * er[i * 4 + 3]) + wwm * ad[i * 4 + 3];
      op[i] = o;
      acc[i * 4 + 0] += wrm * v.x;
      acc[i * 4 + 1] += wrm * v.y;
      acc[i * 4 + 2] += wrm * v.z;
      acc[i * 4 + 3] += wrm * v.w;
    }
  }
  __shared__ float red[256 * 16];
#pragma unroll
  for (int i = 0; i < 16; ++i) red[tid * 16 + i] = acc[i];
  __syncthreads();
  for (int s = 128; s >= 4; s >>= 1) {   // strides stay multiples of 4 -> per-colgroup sums
    if (tid < s) {
#pragma unroll
      for (int i = 0; i < 16; ++i) red[tid * 16 + i] += red[(tid + s) * 16 + i];
    }
    __syncthreads();
  }
  if (tid < 4) {
    float* rp = ws + OFF_RPART + ((size_t)b * K2_BLKS + chunk) * D + tid * 16;
#pragma unroll
    for (int i = 0; i < 16; ++i) rp[i] = red[tid * 16 + i];
  }
}

// ---------------- K6: deterministic read-content reduction ----------------
__global__ __launch_bounds__(64) void k6_reduce_read(const float* __restrict__ ws,
                                                     float* __restrict__ out_read) {
  int b = blockIdx.x, d = threadIdx.x;
  float a = 0.f;
  for (int blk = 0; blk < K2_BLKS; ++blk)
    a += ws[OFF_RPART + ((size_t)b * K2_BLKS + blk) * D + d];
  out_read[(size_t)b * D + d] = a;
}

extern "C" void kernel_launch(void* const* d_in, const int* in_sizes, int n_in,
                              void* d_out, int out_size, void* d_ws, size_t ws_size,
                              hipStream_t stream) {
  const float* mem    = (const float*)d_in[0];
  const float* ctrl   = (const float*)d_in[1];
  const float* Wk_r   = (const float*)d_in[2];
  const float* Wb_r   = (const float*)d_in[3];
  const float* bb_r   = (const float*)d_in[4];
  const float* Wg_r   = (const float*)d_in[5];
  const float* bg_r   = (const float*)d_in[6];
  const float* Ws_r   = (const float*)d_in[7];
  const float* bs_r   = (const float*)d_in[8];
  const float* Wgam_r = (const float*)d_in[9];
  const float* bgam_r = (const float*)d_in[10];
  const float* init_r = (const float*)d_in[11];
  const float* Wk_w   = (const float*)d_in[12];
  const float* Wb_w   = (const float*)d_in[13];
  const float* bb_w   = (const float*)d_in[14];
  const float* Wg_w   = (const float*)d_in[15];
  const float* bg_w   = (const float*)d_in[16];
  const float* Ws_w   = (const float*)d_in[17];
  const float* bs_w   = (const float*)d_in[18];
  const float* Wgam_w = (const float*)d_in[19];
  const float* bgam_w = (const float*)d_in[20];
  const float* init_w = (const float*)d_in[21];
  const float* We     = (const float*)d_in[22];
  const float* be     = (const float*)d_in[23];
  const float* Wa     = (const float*)d_in[24];
  const float* ba     = (const float*)d_in[25];

  float* ws       = (float*)d_ws;
  float* out_read = (float*)d_out;
  float* out_mem  = (float*)d_out + (size_t)B * D;

  k1_small<<<B, 256, 0, stream>>>(ctrl,
      Wk_r, Wb_r, bb_r, Wg_r, bg_r, Ws_r, bs_r, Wgam_r, bgam_r,
      Wk_w, Wb_w, bb_w, Wg_w, bg_w, Ws_w, bs_w, Wgam_w, bgam_w,
      We, be, Wa, ba, ws);
  k2_sim<<<dim3(K2_BLKS, B), 256, 0, stream>>>(mem, ws);
  k3_stats<<<B, 256, 0, stream>>>(init_r, init_w, ws);
  k5_final<<<dim3(K2_BLKS, B), 256, 0, stream>>>(mem, init_r, init_w, ws, out_mem);
  k6_reduce_read<<<B, 64, 0, stream>>>(ws, out_read);
}
